// TimeDecayRichAttention_49039936586046
// MI455X (gfx1250) — matrix operand order repaired and checked by reading, not compiled, not run on hardware
//
#include <hip/hip_runtime.h>
#include <hip/hip_bf16.h>

// TimeDecayRichAttention for MI455X (gfx1250, wave32, WMMA).
// Layer-1 restructured: x@W1 = keys @ (W1a + W1d + diag(q)W1c) + bias(q).
// Layer-1 AND layer-2 run on v_wmma_f32_16x16x32_bf16 (fp32 accumulate);
// layer-3 dot, softmax, and w@keys stay fp32. Kernel is then HBM-bound
// (2 x 210 MB keys traffic at 23.3 TB/s ~ 18 us).
// LDS pool ~139 KB with region aliasing -> 2 workgroups per WGP (320 KB).

typedef __attribute__((ext_vector_type(16))) __bf16 v16bf;
typedef __attribute__((ext_vector_type(8)))  float  v8f;

union ABFrag { v16bf v; unsigned int u[8]; };

#define BB 2048
#define SS 200
#define DD 128
#define H1N 64
#define H2N 32
#define MT 13              // 13*16 = 208 >= 200 rows (padded)

// ---- shared-memory pool layout (bytes, 16-aligned) ----------------------
#define OFF_R0    0                         // kbf u32[208][68]  (56,576) -> h1bf u32[208][36]
#define OFF_R1    56576                     // w1u u16[128][72]  (18,432)
#define OFF_R2    75008                     // h1s f32[208][65]  (54,080) -> h2s f32[208][33]
#define OFF_W2U   129088                    // w2u u16[64][40]   ( 5,120)
#define OFF_QS    134208                    // f32[128]
#define OFF_BIAS  134720                    // f32[64]
#define OFF_B2    134976                    // f32[32]
#define OFF_W3    135104                    // f32[32]
#define OFF_SC    135232                    // f32[256]
#define OFF_RED   136256                    // f32[256]
#define OFF_WS    137280                    // f32[256]
#define OFF_PSUM  138304                    // f32[256]
#define SMEM_SZ   139328

__device__ __forceinline__ unsigned short f2bf(float f) {
    unsigned int u = __float_as_uint(f);
    unsigned int r = (u + 0x7FFFu + ((u >> 16) & 1u)) >> 16;   // RNE
    return (unsigned short)r;
}
__device__ __forceinline__ unsigned int pack2bf(float a, float b) {
    return (unsigned int)f2bf(a) | ((unsigned int)f2bf(b) << 16);
}

__global__ __launch_bounds__(256) void
din_attn_kernel(const float* __restrict__ qry, const float* __restrict__ keys,
                const unsigned char* __restrict__ kmask,
                const float* __restrict__ W1, const float* __restrict__ b1,
                const float* __restrict__ a1p,
                const float* __restrict__ W2, const float* __restrict__ b2,
                const float* __restrict__ a2p,
                const float* __restrict__ W3, const float* __restrict__ b3p,
                float* __restrict__ out_sum, float* __restrict__ out_w) {
    __shared__ __align__(16) unsigned char smem[SMEM_SZ];

    auto* kbf  = reinterpret_cast<unsigned int(*)[68]>(smem + OFF_R0);   // keys bf16 pairs
    auto* h1bf = reinterpret_cast<unsigned int(*)[36]>(smem + OFF_R0);   // aliases kbf
    auto* w1u  = reinterpret_cast<unsigned short(*)[72]>(smem + OFF_R1); // W1eff bf16 [K=128][N=64]
    auto* h1s  = reinterpret_cast<float(*)[65]>(smem + OFF_R2);          // layer-1 out fp32
    auto* h2s  = reinterpret_cast<float(*)[33]>(smem + OFF_R2);          // aliases h1s
    auto* w2u  = reinterpret_cast<unsigned short(*)[40]>(smem + OFF_W2U);// W2 bf16 [K=64][N=32]
    float* qs      = reinterpret_cast<float*>(smem + OFF_QS);
    float* bias1s  = reinterpret_cast<float*>(smem + OFF_BIAS);
    float* b2s     = reinterpret_cast<float*>(smem + OFF_B2);
    float* w3s     = reinterpret_cast<float*>(smem + OFF_W3);
    float* scoress = reinterpret_cast<float*>(smem + OFF_SC);
    float* red     = reinterpret_cast<float*>(smem + OFF_RED);
    float* ws      = reinterpret_cast<float*>(smem + OFF_WS);
    float* psum    = reinterpret_cast<float*>(smem + OFF_PSUM);

    const int tid = threadIdx.x;
    const int b   = blockIdx.x;
    const float a1v = a1p[0];
    const float a2v = a2p[0];
    const float b3v = b3p[0];

    const int wvid = tid >> 5;
    const int lane = tid & 31;
    const int hh   = lane >> 4;            // lane half
    const int ln   = lane & 15;

    // ---- phase A: query row to LDS -------------------------------------
    if (tid < DD) qs[tid] = qry[(size_t)b * DD + tid];
    __syncthreads();

    // ---- phase B: stage keys (bf16), per-b effective W1, bias, W2/W3 ---
    {
        const float* kb = keys + (size_t)b * SS * DD;
        for (int e = tid; e < MT * 16 * 64; e += 256) {
            int row = e >> 6, c = e & 63;
            unsigned int pk = 0u;
            if (row < SS)
                pk = pack2bf(kb[row * DD + 2 * c], kb[row * DD + 2 * c + 1]);
            kbf[row][c] = pk;
        }
    }
    for (int e = tid; e < DD * H1N; e += 256) {        // W1eff = W1a + W1d + q*W1c
        int d = e >> 6, hc = e & 63;
        float wv = W1[d * H1N + hc] + W1[(384 + d) * H1N + hc]
                 + qs[d] * W1[(256 + d) * H1N + hc];
        w1u[d][hc] = f2bf(wv);
    }
    if (tid < H1N) {                                   // bias1 = b1 + q @ (W1b - W1d)
        float s = b1[tid];
        for (int d = 0; d < DD; ++d)
            s += qs[d] * (W1[(128 + d) * H1N + tid] - W1[(384 + d) * H1N + tid]);
        bias1s[tid] = s;
    }
    for (int e = tid; e < H1N * H2N; e += 256)         // W2 -> bf16 B layout
        w2u[e >> 5][e & 31] = f2bf(W2[e]);
    if (tid < H2N) { b2s[tid] = b2[tid]; w3s[tid] = W3[tid]; }
    __syncthreads();

    // ---- phase C: layer-1 GEMM via WMMA bf16 (fp32 accumulate) ---------
    for (int t = wvid; t < MT * 4; t += 8) {           // 13 M-tiles x 4 N-tiles
        const int mt = t >> 2, nt = t & 3;
        v8f acc = {};
        const int ncol = nt * 16 + ln;
        const int arow = mt * 16 + ln;
        #pragma unroll
        for (int kt = 0; kt < 4; ++kt) {
            ABFrag A, Bf;
            #pragma unroll
            for (int j = 0; j < 8; ++j) {              // A 16x32: K = 8h+2j / 16+8h+2(j-4)
                int uidx = kt * 16 + 4 * hh + (j < 4 ? j : 4 + j);
                A.u[j] = kbf[arow][uidx];
            }
            #pragma unroll
            for (int j = 0; j < 8; ++j) {              // B 32x16: K = 8h+2j / 16+8h+2(j-4), N = ln
                int kk = kt * 32 + 8 * hh + (j < 4 ? 2 * j : 8 + 2 * j);
                Bf.u[j] = (unsigned int)w1u[kk][ncol]
                        | ((unsigned int)w1u[kk + 1][ncol] << 16);
            }
            acc = __builtin_amdgcn_wmma_f32_16x16x32_bf16(
                false, A.v, false, Bf.v, (short)0, acc, false, false);
        }
        const float bn = bias1s[ncol];
        #pragma unroll
        for (int r = 0; r < 8; ++r) {                  // C/D: M = r + 8h, N = ln
            int m = mt * 16 + 8 * hh + r;
            float x = acc[r] + bn;
            h1s[m][ncol] = fmaxf(x, 0.f) + a1v * fminf(x, 0.f);   // PReLU
        }
    }
    __syncthreads();

    // ---- phase C2: pack h1 -> bf16 pairs (overwrites dead kbf region) --
    for (int e = tid; e < MT * 16 * 32; e += 256) {
        int row = e >> 5, c = e & 31;
        h1bf[row][c] = pack2bf(h1s[row][2 * c], h1s[row][2 * c + 1]);
    }
    __syncthreads();

    // ---- phase C3: layer-2 GEMM via WMMA bf16, bias+PReLU epilogue -----
    for (int t = wvid; t < MT * 2; t += 8) {           // 13 M-tiles x 2 N-tiles
        const int mt = t >> 1, nt = t & 1;
        v8f acc = {};
        const int ncol = nt * 16 + ln;                 // n < 32
        const int arow = mt * 16 + ln;
        #pragma unroll
        for (int kt = 0; kt < 2; ++kt) {               // K = 64 -> 2 steps of 32
            ABFrag A, Bf;
            #pragma unroll
            for (int j = 0; j < 8; ++j) {
                int uidx = kt * 16 + 4 * hh + (j < 4 ? j : 4 + j);
                A.u[j] = h1bf[arow][uidx];
            }
            #pragma unroll
            for (int j = 0; j < 8; ++j) {
                int kk = kt * 32 + 8 * hh + (j < 4 ? 2 * j : 8 + 2 * j);
                Bf.u[j] = (unsigned int)w2u[kk][ncol]
                        | ((unsigned int)w2u[kk + 1][ncol] << 16);
            }
            acc = __builtin_amdgcn_wmma_f32_16x16x32_bf16(
                false, A.v, false, Bf.v, (short)0, acc, false, false);
        }
        const float bn = b2s[ncol];
        #pragma unroll
        for (int r = 0; r < 8; ++r) {                  // writes alias dead h1 fp32
            int m = mt * 16 + 8 * hh + r;
            float x = acc[r] + bn;
            h2s[m][ncol] = fmaxf(x, 0.f) + a2v * fminf(x, 0.f);   // PReLU
        }
    }
    __syncthreads();

    // ---- phase D: layer-3 dot (32 MACs), time decay, mask --------------
    if (tid < SS) {
        float sc = b3v;
        #pragma unroll
        for (int c = 0; c < H2N; ++c) sc += h2s[tid][c] * w3s[c];
        sc *= __expf(0.1f * (float)(tid - (SS - 1)));           // time decay
        bool valid = kmask[(size_t)b * SS + tid] != 0;
        scoress[tid] = valid ? sc : -INFINITY;
    } else {
        scoress[tid] = -INFINITY;
    }
    __syncthreads();

    // ---- phase E: masked softmax (max + sum tree reductions) -----------
    red[tid] = scoress[tid];
    __syncthreads();
    for (int off = 128; off > 0; off >>= 1) {
        if (tid < off) red[tid] = fmaxf(red[tid], red[tid + off]);
        __syncthreads();
    }
    const float maxv = red[0];
    __syncthreads();
    const float scv = scoress[tid];
    const float ev  = (scv == -INFINITY) ? 0.f : __expf(scv - maxv);
    red[tid] = ev;
    __syncthreads();
    for (int off = 128; off > 0; off >>= 1) {
        if (tid < off) red[tid] += red[tid + off];
        __syncthreads();
    }
    const float sumv = red[0];
    const float wv = (sumv > 0.f) ? ev / sumv : 0.f;            // all-masked -> 0
    ws[tid] = wv;
    if (tid < SS) out_w[(size_t)b * SS + tid] = wv;
    __syncthreads();

    // ---- phase F: weighted sum  out[d] = sum_s w[s]*keys[s][d] ---------
    {
        const int d    = tid & (DD - 1);
        const int half = tid >> 7;
        const float* kb = keys + (size_t)b * SS * DD;
        float p = 0.f;
        const int s0 = half * 100, s1 = s0 + 100;
        for (int s = s0; s < s1; ++s)
            p += ws[s] * kb[s * DD + d];                         // coalesced, L2-hot
        psum[tid] = p;
    }
    __syncthreads();
    if (tid < DD)
        out_sum[(size_t)b * DD + tid] = psum[tid] + psum[tid + DD];
}

extern "C" void kernel_launch(void* const* d_in, const int* in_sizes, int n_in,
                              void* d_out, int out_size, void* d_ws, size_t ws_size,
                              hipStream_t stream) {
    (void)in_sizes; (void)n_in; (void)d_ws; (void)ws_size; (void)out_size;
    const float* qry   = (const float*)d_in[0];
    const float* keys  = (const float*)d_in[1];
    const unsigned char* kmask = (const unsigned char*)d_in[2];   // jnp.bool_ = 1 byte
    const float* W1 = (const float*)d_in[3];
    const float* b1 = (const float*)d_in[4];
    const float* a1 = (const float*)d_in[5];
    const float* W2 = (const float*)d_in[6];
    const float* b2 = (const float*)d_in[7];
    const float* a2 = (const float*)d_in[8];
    const float* W3 = (const float*)d_in[9];
    const float* b3 = (const float*)d_in[10];

    float* out_sum = (float*)d_out;                    // [B, D]
    float* out_w   = out_sum + (size_t)BB * DD;        // [B, S]

    din_attn_kernel<<<BB, 256, 0, stream>>>(qry, keys, kmask, W1, b1, a1,
                                            W2, b2, a2, W3, b3, out_sum, out_w);
}
